// re3_87505663689473
// MI455X (gfx1250) — matrix operand order repaired and checked by reading, not compiled, not run on hardware
//
#include <hip/hip_runtime.h>

// ---------------------------------------------------------------------------
// Problem constants (from reference): N=16384 rows, D=128, H=L=256, k=3
// ---------------------------------------------------------------------------
#define NROW 16384
#define DIN  128
#define HID  256

typedef unsigned short u16t;
typedef __attribute__((ext_vector_type(16))) __bf16 v16bf;
typedef __attribute__((ext_vector_type(8)))  __bf16 v8bf;
typedef __attribute__((ext_vector_type(8)))  float  v8f;

// LDS padded strides (elements). Chosen so row/col bases land on distinct banks.
#define ACT_STRIDE 264   // u16 per activation row (256 + 8)
#define WT_STRIDE  264   // u16 per transposed-weight column (max K=256 + 8)
#define Z_STRIDE   257   // f32 per z row (257 % 64 == 1 -> conflict-free column reads)

__device__ __forceinline__ u16t f2bf(float f) {
    // round-to-nearest-even fp32 -> bf16
    unsigned int u = __float_as_uint(f);
    unsigned int r = u + 0x7FFFu + ((u >> 16) & 1u);
    return (u16t)(r >> 16);
}

__device__ __forceinline__ v16bf mk16(v8bf lo, v8bf hi) {
    v16bf r;
#pragma unroll
    for (int i = 0; i < 8; ++i) { r[i] = lo[i]; r[i + 8] = hi[i]; }
    return r;
}

// Branchless insert of x into sorted triple (a <= b <= c), keeping 3 smallest.
__device__ __forceinline__ void ins3(float x, float& a, float& b, float& c) {
    const float s0 = fminf(x, a);
    const float m0 = fmaxf(x, a);
    const float s1 = fminf(m0, b);
    const float m1 = fmaxf(m0, b);
    const float s2 = fminf(m1, c);
    a = s0; b = s1; c = s2;
}

// ---------------------------------------------------------------------------
// One MLP layer for a wave's 16-row tile: actIn[16 x KIN] (bf16, LDS) times
// wT[256 x KIN] (bf16, LDS, transposed) -> 16 x 256 output.
// RELU=true: bias+relu -> bf16 actOut.  RELU=false: bias -> f32 zOut.
// ---------------------------------------------------------------------------
template <int KIN, bool RELU>
__device__ __forceinline__ void run_layer(const u16t* __restrict__ actIn,
                                          const u16t* __restrict__ wT,
                                          const float* __restrict__ bias,
                                          u16t* __restrict__ actOut,
                                          float* __restrict__ zOut,
                                          int rb, int lane) {
    const int hl  = (lane < 16) ? 0 : 1;
    const int l15 = lane & 15;
    constexpr int NK = KIN / 32;

    // Hoist A fragments (depend only on kk): lane carries row rb + l15.
    v16bf afr[NK];
    const u16t* arow = actIn + (rb + l15) * ACT_STRIDE;
#pragma unroll
    for (int kk = 0; kk < NK; ++kk) {
        const int s0 = kk * 32 + hl * 8;
        v8bf a0 = *(const v8bf*)(arow + s0);
        v8bf a1 = *(const v8bf*)(arow + s0 + 16);
        afr[kk] = mk16(a0, a1);
    }

#pragma unroll 1
    for (int n = 0; n < 16; ++n) {
        const int col = n * 16 + l15;
        const float bv = bias[col];
        v8f acc = {};
        const u16t* wcol = wT + col * WT_STRIDE + hl * 8;
#pragma unroll
        for (int kk = 0; kk < NK; ++kk) {
            v8bf b0 = *(const v8bf*)(wcol + kk * 32);
            v8bf b1 = *(const v8bf*)(wcol + kk * 32 + 16);
            acc = __builtin_amdgcn_wmma_f32_16x16x32_bf16(
                false, afr[kk], false, mk16(b0, b1), (short)0, acc, false, false);
        }
#pragma unroll
        for (int v = 0; v < 8; ++v) {
            const int row = rb + v + hl * 8;
            float val = acc[v] + bv;
            if (RELU) {
                val = fmaxf(val, 0.0f);
                actOut[row * ACT_STRIDE + col] = f2bf(val);
            } else {
                zOut[row * Z_STRIDE + col] = val;
            }
        }
    }
}

// ---------------------------------------------------------------------------
// Kernel 1: fused 3-layer MLP + LayerNorm for src (blocks [0,256)) and tgt
// (blocks [256,512)). Block = 128 threads (4 waves), 64 rows per block.
// Emits: f32 encoded src (d_out), bf16 encodings + squared norms (workspace).
// ---------------------------------------------------------------------------
__global__ __launch_bounds__(128, 1)
void encode_kernel(const float* __restrict__ xsrc, const float* __restrict__ xtgt,
                   const float* __restrict__ W1, const float* __restrict__ b1,
                   const float* __restrict__ W2, const float* __restrict__ b2,
                   const float* __restrict__ W3, const float* __restrict__ b3,
                   const float* __restrict__ gamma, const float* __restrict__ beta,
                   float* __restrict__ out_src,
                   u16t* __restrict__ enc_src, u16t* __restrict__ enc_tgt,
                   float* __restrict__ ssq, float* __restrict__ tsq) {
    extern __shared__ char smem[];
    u16t*  actA = (u16t*)smem;                       //  64 x ACT_STRIDE bf16
    u16t*  actB = actA + 64 * ACT_STRIDE;            //  64 x ACT_STRIDE bf16
    u16t*  wT   = actB + 64 * ACT_STRIDE;            // 256 x WT_STRIDE  bf16 (transposed)
    float* zbuf = (float*)(wT + 256 * WT_STRIDE);    //  64 x Z_STRIDE   f32

    const int  bid      = blockIdx.x;
    const bool isSrc    = bid < (NROW / 64);
    const int  rowstart = (isSrc ? bid : bid - (NROW / 64)) * 64;
    const float* x   = isSrc ? xsrc : xtgt;
    u16t*        enc = isSrc ? enc_src : enc_tgt;
    float*       sqn = isSrc ? ssq : tsq;

    const int tid  = threadIdx.x;
    const int lane = tid & 31;
    const int wave = tid >> 5;
    const int rb   = wave * 16;

    // Stage x tile (64 x 128) as bf16.
    for (int idx = tid; idx < 64 * DIN; idx += 128) {
        const int r = idx / DIN, c = idx - r * DIN;
        actA[r * ACT_STRIDE + c] = f2bf(x[(size_t)(rowstart + r) * DIN + c]);
    }
    // Stage W1 transposed (coalesced global reads along the 256-col axis).
    for (int idx = tid; idx < DIN * HID; idx += 128) {
        const int k = idx / HID, c = idx - k * HID;
        wT[c * WT_STRIDE + k] = f2bf(W1[idx]);
    }
    __syncthreads();
    run_layer<DIN, true>(actA, wT, b1, actB, nullptr, rb, lane);
    __syncthreads();

    for (int idx = tid; idx < HID * HID; idx += 128) {
        const int k = idx / HID, c = idx - k * HID;
        wT[c * WT_STRIDE + k] = f2bf(W2[idx]);
    }
    __syncthreads();
    run_layer<HID, true>(actB, wT, b2, actA, nullptr, rb, lane);
    __syncthreads();

    for (int idx = tid; idx < HID * HID; idx += 128) {
        const int k = idx / HID, c = idx - k * HID;
        wT[c * WT_STRIDE + k] = f2bf(W3[idx]);
    }
    __syncthreads();
    run_layer<HID, false>(actA, wT, b3, nullptr, zbuf, rb, lane);
    __syncthreads();

    // LayerNorm: one thread per row (threads 0..63).
    if (tid < 64) {
        const float* zr = zbuf + tid * Z_STRIDE;
        float mu = 0.0f;
#pragma unroll 4
        for (int c = 0; c < HID; ++c) mu += zr[c];
        mu *= (1.0f / HID);
        float var = 0.0f;
#pragma unroll 4
        for (int c = 0; c < HID; ++c) { const float d = zr[c] - mu; var = fmaf(d, d, var); }
        var *= (1.0f / HID);
        const float rstd = rsqrtf(var + 1e-5f);
        const size_t grow = (size_t)(rowstart + tid);
        float sq = 0.0f;
#pragma unroll 4
        for (int c = 0; c < HID; ++c) {
            const float val = (zr[c] - mu) * rstd * gamma[c] + beta[c];
            if (isSrc) out_src[grow * HID + c] = val;
            enc[grow * HID + c] = f2bf(val);
            sq = fmaf(val, val, sq);
        }
        sqn[grow] = sq;
    }
}

// ---------------------------------------------------------------------------
// Kernel 2: fused distance GEMM + top-3 kNN.
//   grid = 64 blocks x 256 threads (8 waves); block owns 256 src rows
//   (32 per wave -> two resident A-fragment sets, 128 VGPRs).
//   Mainloop: stage 128 tgt rows (64 KB bf16) to LDS with coalesced b128
//   loads + prefetch of the next tile, then every B-fragment ds_load feeds
//   TWO WMMAs (1:1 ds:wmma). Per-lane branchless top-3 of squared distances,
//   merged across lanes through LDS (aliased over the tgt tile buffer).
// ---------------------------------------------------------------------------
#define KNN_MB  256         // src rows per block
#define KNN_NT  128         // tgt rows per LDS tile
#define KNN_LST 264         // u16 stride of LDS tgt tile rows

__global__ __launch_bounds__(256, 1)
void knn_kernel(const u16t* __restrict__ encS, const u16t* __restrict__ encT,
                const float* __restrict__ ssq, const float* __restrict__ tsq,
                float* __restrict__ entropy_out) {
    extern __shared__ char smem[];
    u16t*  tlds = (u16t*)smem;          // KNN_NT x KNN_LST bf16 (67584 B)
    float* mbuf = (float*)smem;         // aliased: 256 rows x 16 slots x 3 f32

    const int tid  = threadIdx.x;
    const int lane = tid & 31;
    const int wave = tid >> 5;
    const int hl   = (lane < 16) ? 0 : 1;
    const int l15  = lane & 15;
    const int i0   = blockIdx.x * KNN_MB + wave * 32;

    // Hoist both 16-row A-fragment sets (K = 0..255) and squared norms.
    v16bf afr0[8], afr1[8];
    {
        const u16t* ar0 = encS + (size_t)(i0 + l15) * HID;
        const u16t* ar1 = encS + (size_t)(i0 + 16 + l15) * HID;
#pragma unroll
        for (int kk = 0; kk < 8; ++kk) {
            const int s0 = kk * 32 + hl * 8;
            afr0[kk] = mk16(*(const v8bf*)(ar0 + s0), *(const v8bf*)(ar0 + s0 + 16));
            afr1[kk] = mk16(*(const v8bf*)(ar1 + s0), *(const v8bf*)(ar1 + s0 + 16));
        }
    }
    float snA[8], snB[8];
#pragma unroll
    for (int v = 0; v < 8; ++v) {
        snA[v] = ssq[i0 + v + hl * 8];
        snB[v] = ssq[i0 + 16 + v + hl * 8];
    }

    float t0a[8], t1a[8], t2a[8], t0b[8], t1b[8], t2b[8];
#pragma unroll
    for (int v = 0; v < 8; ++v) {
        t0a[v] = t1a[v] = t2a[v] = 3.0e38f;
        t0b[v] = t1b[v] = t2b[v] = 3.0e38f;
    }

#pragma unroll 1
    for (int j0 = 0; j0 < NROW; j0 += KNN_NT) {
        __syncthreads();   // previous tile fully consumed
        // Cooperative coalesced staging: 128 rows x 512 B = 4096 16-byte chunks.
#pragma unroll 1
        for (int c = tid; c < KNN_NT * 32; c += 256) {
            const int r = c >> 5;
            const int o = (c & 31) * 8;
            *(v8bf*)(tlds + r * KNN_LST + o) =
                *(const v8bf*)(encT + (size_t)(j0 + r) * HID + o);
        }
        // Prefetch next tile into cache while this one is computed (2 lines/thread).
        if (j0 + KNN_NT < NROW) {
            const u16t* nxt = encT + (size_t)(j0 + KNN_NT) * HID;
            __builtin_prefetch(nxt + (size_t)tid * 64, 0, 3);
            __builtin_prefetch(nxt + (size_t)(tid + 256) * 64, 0, 3);
        }
        __syncthreads();

#pragma unroll 1
        for (int jj = 0; jj < KNN_NT / 16; ++jj) {
            const float tn = tsq[j0 + jj * 16 + l15];
            const u16t* trow = tlds + (jj * 16 + l15) * KNN_LST + hl * 8;
            v8f acc0 = {}, acc1 = {};
#pragma unroll
            for (int kk = 0; kk < 8; ++kk) {
                v8bf b0 = *(const v8bf*)(trow + kk * 32);
                v8bf b1 = *(const v8bf*)(trow + kk * 32 + 16);
                const v16bf bf = mk16(b0, b1);
                acc0 = __builtin_amdgcn_wmma_f32_16x16x32_bf16(
                    false, afr0[kk], false, bf, (short)0, acc0, false, false);
                acc1 = __builtin_amdgcn_wmma_f32_16x16x32_bf16(
                    false, afr1[kk], false, bf, (short)0, acc1, false, false);
            }
#pragma unroll
            for (int v = 0; v < 8; ++v) {
                const float d2a = fmaf(-2.0f, acc0[v], snA[v] + tn);
                ins3(d2a, t0a[v], t1a[v], t2a[v]);
                const float d2b = fmaf(-2.0f, acc1[v], snB[v] + tn);
                ins3(d2b, t0b[v], t1b[v], t2b[v]);
            }
        }
    }

    // Cross-lane merge through LDS (aliases the tgt tile buffer).
    __syncthreads();
#pragma unroll
    for (int v = 0; v < 8; ++v) {
        const int rowA = wave * 32 + v + hl * 8;
        const int rowB = rowA + 16;
        float* pa = &mbuf[(rowA * 16 + l15) * 3];
        pa[0] = t0a[v]; pa[1] = t1a[v]; pa[2] = t2a[v];
        float* pb = &mbuf[(rowB * 16 + l15) * 3];
        pb[0] = t0b[v]; pb[1] = t1b[v]; pb[2] = t2b[v];
    }
    __syncthreads();

    {   // one thread per src row: reduce 16 partial top-3 lists -> entropy
        const float* p = &mbuf[(tid * 16) * 3];
        float a = 3.0e38f, b = 3.0e38f, c = 3.0e38f;
#pragma unroll 4
        for (int s = 0; s < 48; ++s) ins3(p[s], a, b, c);
        const float e = 0.5f * (sqrtf(fmaxf(a, 0.0f)) +
                                sqrtf(fmaxf(b, 0.0f)) +
                                sqrtf(fmaxf(c, 0.0f)));
        entropy_out[blockIdx.x * KNN_MB + tid] = e;
    }
}

// ---------------------------------------------------------------------------
// Host launcher
// ---------------------------------------------------------------------------
extern "C" void kernel_launch(void* const* d_in, const int* in_sizes, int n_in,
                              void* d_out, int out_size, void* d_ws, size_t ws_size,
                              hipStream_t stream) {
    const float* xsrc  = (const float*)d_in[0];
    const float* xtgt  = (const float*)d_in[1];
    const float* W1    = (const float*)d_in[2];
    const float* b1    = (const float*)d_in[3];
    const float* W2    = (const float*)d_in[4];
    const float* b2    = (const float*)d_in[5];
    const float* W3    = (const float*)d_in[6];
    const float* b3    = (const float*)d_in[7];
    const float* gamma = (const float*)d_in[8];
    const float* beta  = (const float*)d_in[9];

    float* out = (float*)d_out;            // [N*256] src_enc fp32, then [N] entropy

    // Workspace layout: bf16 encodings + squared norms (~16.9 MB total)
    u16t*  encS = (u16t*)d_ws;
    u16t*  encT = encS + (size_t)NROW * HID;
    float* ssq  = (float*)(encT + (size_t)NROW * HID);
    float* tsq  = ssq + NROW;

    // Encoder dynamic LDS: 2 act buffers + transposed weights + z buffer = 268544 B
    const size_t enc_lds = (size_t)(2 * 64 * ACT_STRIDE + 256 * WT_STRIDE) * sizeof(u16t)
                         + (size_t)64 * Z_STRIDE * sizeof(float);
    encode_kernel<<<2 * (NROW / 64), 128, enc_lds, stream>>>(
        xsrc, xtgt, W1, b1, W2, b2, W3, b3, gamma, beta,
        out, encS, encT, ssq, tsq);

    // kNN dynamic LDS: max(tgt tile, merge buffer) = 128*264*2 = 67584 B
    const size_t knn_lds = (size_t)KNN_NT * KNN_LST * sizeof(u16t);
    knn_kernel<<<NROW / KNN_MB, 256, knn_lds, stream>>>(
        encS, encT, ssq, tsq, out + (size_t)NROW * HID);
}
